// SparseLinear_16028817949059
// MI455X (gfx1250) — matrix operand order repaired and checked by reading, not compiled, not run on hardware
//
#include <hip/hip_runtime.h>

// out = x @ W^T + bias      x:[M,4096] f32, W:[4096,4096] f32, bias:[4096]
// bf16 WMMA GEMM, fp32 accumulate.
// Block = 256 threads (8 waves, 2x4) -> 128x256 block tile; wave tile 64x64.
// K-slabs of 32 staged global->LDS by the Tensor Data Mover (TDM), double
// buffered; wave 0 drives the TDM, barrier publishes the buffer.

#define KDIM 4096
#define NDIM 4096

#define LDS_STRIDE 36              // 32 data floats + 4 pad (bank skew, 16B align)
#define A_ROWS 128
#define B_ROWS 256
#define A_FLOATS (A_ROWS * LDS_STRIDE)       // 4608
#define B_FLOATS (B_ROWS * LDS_STRIDE)       // 9216
#define BUF_FLOATS (A_FLOATS + B_FLOATS)     // 13824 floats = 55296 B per buffer

typedef __attribute__((ext_vector_type(16))) __bf16       v16bf;
typedef __attribute__((ext_vector_type(8)))  float        v8f;
typedef __attribute__((ext_vector_type(4)))  float        f32x4;
typedef __attribute__((ext_vector_type(4)))  unsigned int u32x4;
typedef __attribute__((ext_vector_type(4)))  int          i32x4;
typedef __attribute__((ext_vector_type(8)))  int          i32x8;

// Issue one 2D TDM tile load: rows x 36 fp32 elements, tensor row stride 4096.
// tensor_dim0 = remaining K (so the 4 pad elements past K read as zero),
// tensor_dim1 = remaining rows. Descriptor layout per CDNA5 ISA §8.3/8.4.
__device__ __forceinline__ void tdm_load_tile(unsigned lds_off_bytes,
                                              const float* gsrc,
                                              unsigned remK,
                                              unsigned remRows,
                                              unsigned tileRows)
{
    unsigned long long ga = (unsigned long long)(uintptr_t)gsrc;
    u32x4 g0;
    g0[0] = 1u;                                        // count=1, user mode, no gather
    g0[1] = lds_off_bytes;                             // lds_addr  (bits 63:32)
    g0[2] = (unsigned)ga;                              // global_addr[31:0]
    g0[3] = (unsigned)((ga >> 32) & 0x01FFFFFFu)       // global_addr[56:32]
          | (2u << 30);                                // type = 2
    i32x8 g1;
    g1[0] = (int)(2u << 16);                           // data_size = 2 (4 bytes)
    g1[1] = (int)((remK & 0xFFFFu) << 16);             // tensor_dim0[15:0]
    g1[2] = (int)(((remK >> 16) & 0xFFFFu)             // tensor_dim0[31:16]
          | ((remRows & 0xFFFFu) << 16));              // tensor_dim1[15:0]
    g1[3] = (int)(((remRows >> 16) & 0xFFFFu)          // tensor_dim1[31:16]
          | ((unsigned)LDS_STRIDE << 16));             // tile_dim0 = 36
    g1[4] = (int)(tileRows & 0xFFFFu);                 // tile_dim1 (tile_dim2 = 0)
    g1[5] = (int)KDIM;                                 // tensor_dim0_stride[31:0]
    g1[6] = 0;                                         // stride0 hi / stride1 lo
    g1[7] = 0;
    i32x4 z4 = (i32x4){0, 0, 0, 0};                    // groups 2/3 unused (2D)
    i32x8 z8 = (i32x8){0, 0, 0, 0, 0, 0, 0, 0};
    __builtin_amdgcn_tensor_load_to_lds(g0, g1, z4, z4, z8, 0);
}

__global__ __launch_bounds__(256)
void sparse_linear_wmma_kernel(const float* __restrict__ X,
                               const float* __restrict__ W,
                               const float* __restrict__ Bias,
                               float* __restrict__ Out,
                               int M)
{
    __shared__ __attribute__((aligned(16))) float smem[2 * BUF_FLOATS];

    const int lane  = threadIdx.x & 31;
    const int wave  = threadIdx.x >> 5;
    const int waveM = wave >> 2;    // 0..1
    const int waveN = wave & 3;     // 0..3

    const int blockRow = blockIdx.y * 128;   // A rows staged by this block
    const int blockCol = blockIdx.x * 256;   // W rows staged by this block
    const int mBlock = blockRow + waveM * 64;
    const int nBlock = blockCol + waveN * 64;

    // Fragment per-lane geometry (ISA 7.12.2):
    const int rowA = lane & 15;             // A/B row within 16-row tile
    const int kbA  = (lane >> 4) << 3;      // A: K runs [kbA,kbA+8) and +16
    const int kbB  = (lane >> 4) << 3;      // B: K runs [kbB,kbB+8) and +16

    const unsigned smemBase = (unsigned)(uintptr_t)(&smem[0]);  // LDS byte offset

    // ---- accumulators initialized with bias (C/D: N = lane&15 in every VGPR)
    v8f c[4][4];
#pragma unroll
    for (int j = 0; j < 4; ++j) {
        const float bv = Bias[nBlock + j * 16 + (lane & 15)];
        v8f cv;
#pragma unroll
        for (int e = 0; e < 8; ++e) cv[e] = bv;
#pragma unroll
        for (int i = 0; i < 4; ++i) c[i][j] = cv;
    }

    // ---- prologue: stage K-slab 0 into buffer 0
    if (wave == 0) {
        tdm_load_tile(smemBase,
                      X + (size_t)blockRow * KDIM,
                      KDIM, (unsigned)(M - blockRow), A_ROWS);
        tdm_load_tile(smemBase + A_FLOATS * 4,
                      W + (size_t)blockCol * KDIM,
                      KDIM, (unsigned)(NDIM - blockCol), B_ROWS);
        __builtin_amdgcn_s_wait_tensorcnt(0);
    }
    __syncthreads();

    int p = 0;
    for (int k0 = 0; k0 < KDIM; k0 += 32) {
        // ---- prefetch next K-slab into the other buffer via TDM
        if (wave == 0 && (k0 + 32) < KDIM) {
            const unsigned dst = smemBase + (unsigned)((p ^ 1) * BUF_FLOATS) * 4;
            tdm_load_tile(dst,
                          X + (size_t)blockRow * KDIM + (k0 + 32),
                          (unsigned)(KDIM - (k0 + 32)),
                          (unsigned)(M - blockRow), A_ROWS);
            tdm_load_tile(dst + A_FLOATS * 4,
                          W + (size_t)blockCol * KDIM + (k0 + 32),
                          (unsigned)(KDIM - (k0 + 32)),
                          (unsigned)(NDIM - blockCol), B_ROWS);
        }

        const float* bufA = smem + p * BUF_FLOATS;
        const float* bufB = bufA + A_FLOATS;

        // ---- A fragments from LDS (ds_load_b128, 16B aligned, skewed rows)
        v16bf a[4];
#pragma unroll
        for (int i = 0; i < 4; ++i) {
            const float* pa = bufA
                + (size_t)(waveM * 64 + i * 16 + rowA) * LDS_STRIDE + kbA;
            f32x4 x0 = *(const f32x4*)(pa);
            f32x4 x1 = *(const f32x4*)(pa + 4);
            f32x4 x2 = *(const f32x4*)(pa + 16);
            f32x4 x3 = *(const f32x4*)(pa + 20);
            v16bf t;
#pragma unroll
            for (int e = 0; e < 4; ++e) {
                t[e]      = (__bf16)x0[e];
                t[4 + e]  = (__bf16)x1[e];
                t[8 + e]  = (__bf16)x2[e];
                t[12 + e] = (__bf16)x3[e];
            }
            a[i] = t;
        }

        // ---- B fragments: ping-pong two register sets so the next group's
        //      conversions don't WAR-hazard the registers the previous 4 WMMAs
        //      just read (avoids the 4x v_nop co-exec bubbles).
        v16bf bfr[2];
#pragma unroll
        for (int j = 0; j < 4; ++j) {
            const float* pb = bufB
                + (size_t)(waveN * 64 + j * 16 + rowA) * LDS_STRIDE + kbB;
            f32x4 y0 = *(const f32x4*)(pb);
            f32x4 y1 = *(const f32x4*)(pb + 4);
            f32x4 y2 = *(const f32x4*)(pb + 16);
            f32x4 y3 = *(const f32x4*)(pb + 20);
            v16bf t;
#pragma unroll
            for (int e = 0; e < 4; ++e) {
                t[e]      = (__bf16)y0[e];
                t[4 + e]  = (__bf16)y1[e];
                t[8 + e]  = (__bf16)y2[e];
                t[12 + e] = (__bf16)y3[e];
            }
            bfr[j & 1] = t;
#pragma unroll
            for (int i = 0; i < 4; ++i) {
                c[i][j] = __builtin_amdgcn_wmma_f32_16x16x32_bf16(
                    false, a[i], false, bfr[j & 1], (short)0, c[i][j], false, false);
            }
        }

        // ---- publish next buffer: wave0 waits TDM, then everyone barriers
        if (wave == 0) __builtin_amdgcn_s_wait_tensorcnt(0);
        __syncthreads();
        p ^= 1;
    }

    // ---- store (C/D layout: VGPR v, lane L -> M = v + 8*(L>>4), N = L&15)
    const int nCol = lane & 15;
    const int mOff = (lane >> 4) << 3;
#pragma unroll
    for (int i = 0; i < 4; ++i) {
#pragma unroll
        for (int j = 0; j < 4; ++j) {
            float* outp = Out + (size_t)(mBlock + i * 16 + mOff) * NDIM
                              + (nBlock + j * 16 + nCol);
#pragma unroll
            for (int v = 0; v < 8; ++v) {
                outp[(size_t)v * NDIM] = c[i][j][v];
            }
        }
    }
}

extern "C" void kernel_launch(void* const* d_in, const int* in_sizes, int n_in,
                              void* d_out, int out_size, void* d_ws, size_t ws_size,
                              hipStream_t stream) {
    const float* x    = (const float*)d_in[0];   // [M, 4096]
    const float* w    = (const float*)d_in[1];   // [4096, 4096]
    const float* bias = (const float*)d_in[2];   // [4096]
    float*       out  = (float*)d_out;           // [M, 4096]

    const int M = in_sizes[0] / KDIM;            // 8192

    dim3 block(256, 1, 1);
    dim3 grid(NDIM / 256, M / 128, 1);           // (16, 64)
    sparse_linear_wmma_kernel<<<grid, block, 0, stream>>>(x, w, bias, out, M);
}
